// SelfAttention_6889127543258
// MI455X (gfx1250) — matrix operand order repaired and checked by reading, not compiled, not run on hardware
//
#include <hip/hip_runtime.h>
#include <hip/hip_bf16.h>
#include <math.h>

// Problem constants (match reference)
#define B_ 4
#define T_ 2048
#define D_ 512
#define H_ 8
#define C_ 64

typedef __attribute__((ext_vector_type(16))) __bf16 v16bf;
typedef __attribute__((ext_vector_type(8)))  float  v8f;

union Frag16 {
    v16bf v;
    uint4 q[2];
};

static __device__ __forceinline__ v8f v8zero() {
    v8f z;
#pragma unroll
    for (int i = 0; i < 8; ++i) z[i] = 0.0f;
    return z;
}

static __device__ __forceinline__ v8f wmma_bf16(v16bf a, v16bf b, v8f c) {
    // D = A(16x32 bf16) * B(32x16 bf16) + C(16x16 f32)
    return __builtin_amdgcn_wmma_f32_16x16x32_bf16(false, a, false, b, (short)0, c,
                                                   false, false);
}

// ds_swizzle group-of-32 xor pattern: offset = and_mask(0x1F) | (xor_mask << 10)
template <int PAT>
static __device__ __forceinline__ float dswz(float v) {
    return __int_as_float(__builtin_amdgcn_ds_swizzle(__float_as_int(v), PAT));
}
// reduce across the 16-lane half-group (lanes sharing lane>>4)
static __device__ __forceinline__ float redmax16(float v) {
    v = fmaxf(v, dswz<0x041F>(v));   // xor 1
    v = fmaxf(v, dswz<0x081F>(v));   // xor 2
    v = fmaxf(v, dswz<0x101F>(v));   // xor 4
    v = fmaxf(v, dswz<0x201F>(v));   // xor 8
    return v;
}
static __device__ __forceinline__ float redsum16(float v) {
    v += dswz<0x041F>(v);
    v += dswz<0x081F>(v);
    v += dswz<0x101F>(v);
    v += dswz<0x201F>(v);
    return v;
}
static __device__ __forceinline__ float redsum32(float v) {
    v = redsum16(v);
    v += dswz<0x401F>(v);            // xor 16
    return v;
}

// ---------------------------------------------------------------------------
// Kernel 0: cast fp32 weight matrix -> bf16 (D*D elements, grid covers exactly)
// ---------------------------------------------------------------------------
__global__ __launch_bounds__(256) void castw_kernel(const float* __restrict__ src,
                                                    __bf16* __restrict__ dst) {
    int i = blockIdx.x * 256 + threadIdx.x;
    dst[i] = (__bf16)src[i];
}

// ---------------------------------------------------------------------------
// Kernel 1: LayerNorm + cast to bf16.  One wave per row (D=512 -> 16 f32/lane).
// ---------------------------------------------------------------------------
__global__ __launch_bounds__(256) void ln_cast_kernel(const float* __restrict__ x,
                                                      const float* __restrict__ gamma,
                                                      const float* __restrict__ beta,
                                                      __bf16* __restrict__ xn) {
    const int lane = threadIdx.x & 31;
    const int wave = threadIdx.x >> 5;
    const int row  = blockIdx.x * 8 + wave;           // [0, B*T)
    const float* xr = x + (size_t)row * D_ + lane * 16;

    float vals[16];
    const float4* xv = (const float4*)xr;
#pragma unroll
    for (int i = 0; i < 4; ++i) {
        float4 f = xv[i];
        vals[i * 4 + 0] = f.x; vals[i * 4 + 1] = f.y;
        vals[i * 4 + 2] = f.z; vals[i * 4 + 3] = f.w;
    }
    float s = 0.f, sq = 0.f;
#pragma unroll
    for (int i = 0; i < 16; ++i) { s += vals[i]; sq += vals[i] * vals[i]; }
    s  = redsum32(s);
    sq = redsum32(sq);
    const float mean = s * (1.0f / D_);
    const float var  = sq * (1.0f / D_) - mean * mean;
    const float rstd = rsqrtf(var + 1e-5f);

    float g[16], bb[16];
    const float4* gv = (const float4*)(gamma + lane * 16);
    const float4* bv = (const float4*)(beta + lane * 16);
#pragma unroll
    for (int i = 0; i < 4; ++i) {
        float4 f = gv[i];
        g[i * 4 + 0] = f.x; g[i * 4 + 1] = f.y; g[i * 4 + 2] = f.z; g[i * 4 + 3] = f.w;
        float4 e = bv[i];
        bb[i * 4 + 0] = e.x; bb[i * 4 + 1] = e.y; bb[i * 4 + 2] = e.z; bb[i * 4 + 3] = e.w;
    }
    Frag16 o;
#pragma unroll
    for (int i = 0; i < 16; ++i)
        o.v[i] = (__bf16)((vals[i] - mean) * rstd * g[i] + bb[i]);

    uint4* dst = (uint4*)(xn + (size_t)row * D_ + lane * 16);
    dst[0] = o.q[0];
    dst[1] = o.q[1];
}

// ---------------------------------------------------------------------------
// Kernel 2: QKV projection, NT-GEMM via WMMA bf16, register double-buffered.
//   y[t,j] = sum_d xn[t,d]*W[j,d]   (both operands contiguous along d)
//   grid = (B*T/16, 3); block = 256 (8 waves). Wave w -> N cols [w*64, w*64+64).
//   which==2 (V) is stored transposed: vt[(b*D + j)*T + t]
// ---------------------------------------------------------------------------
__global__ __launch_bounds__(256)
void qkv_kernel(const __bf16* __restrict__ xn,
                const __bf16* __restrict__ wqb, const __bf16* __restrict__ wkb,
                const __bf16* __restrict__ wvb,
                __bf16* __restrict__ qo, __bf16* __restrict__ ko,
                __bf16* __restrict__ vt) {
    const int lane  = threadIdx.x & 31;
    const int wave  = threadIdx.x >> 5;
    const int lhalf = lane >> 4;
    const int llo   = lane & 15;
    const int t0    = blockIdx.x * 16;                // row tile in [0, B*T)
    const int which = blockIdx.y;                     // 0:q 1:k 2:v
    const int j0    = wave * 64;

    const __bf16* W = (which == 0) ? wqb : ((which == 1) ? wkb : wvb);
    const __bf16* arow = xn + (size_t)(t0 + llo) * D_;

    v8f acc[4] = {v8zero(), v8zero(), v8zero(), v8zero()};

    auto loadAB = [&](int kb, Frag16& a, v16bf (&bv)[4]) {
        a.q[0] = *(const uint4*)(arow + kb + lhalf * 16);
        a.q[1] = *(const uint4*)(arow + kb + 8 + lhalf * 16);
#pragma unroll
        for (int nt = 0; nt < 4; ++nt) {
            bv[nt] = *(const v16bf*)(W + (size_t)(j0 + nt * 16 + llo) * D_ + kb +
                                     lhalf * 16);
        }
    };

    Frag16 a0, a1;
    v16bf  b0[4], b1[4];
    loadAB(0, a0, b0);

    for (int kb = 0; kb < D_; kb += 64) {
        loadAB(kb + 32, a1, b1);                      // prefetch next half-step
#pragma unroll
        for (int nt = 0; nt < 4; ++nt) acc[nt] = wmma_bf16(a0.v, b0[nt], acc[nt]);
        loadAB((kb + 64) & (D_ - 1), a0, b0);         // prefetch next step (wraps)
#pragma unroll
        for (int nt = 0; nt < 4; ++nt) acc[nt] = wmma_bf16(a1.v, b1[nt], acc[nt]);
    }

    if (which == 2) {
        const int bidx = t0 / T_;
#pragma unroll
        for (int nt = 0; nt < 4; ++nt) {
            const int j = j0 + nt * 16 + llo;
#pragma unroll
            for (int r = 0; r < 8; ++r) {
                const int tl = (t0 % T_) + r + lhalf * 8;
                vt[(size_t)(bidx * D_ + j) * T_ + tl] = (__bf16)acc[nt][r];
            }
        }
    } else {
        __bf16* out = (which == 0) ? qo : ko;
#pragma unroll
        for (int nt = 0; nt < 4; ++nt) {
#pragma unroll
            for (int r = 0; r < 8; ++r) {
                out[(size_t)(t0 + r + lhalf * 8) * D_ + j0 + nt * 16 + llo] =
                    (__bf16)acc[nt][r];
            }
        }
    }
}

// ---------------------------------------------------------------------------
// Kernel 3: fused flash-style attention + residual, double-buffered K/V frags.
//   grid = (B*H, T/128); block = 256 (8 waves); wave -> one 16-query tile.
//   Per 32-key step: 4 WMMA (S = Q*K^T) + online softmax + 4 WMMA (O += P*V).
// ---------------------------------------------------------------------------
struct KVFrags {
    v16bf k[4];   // [subtile(0..1)*2 + kb(0..1)]
    v16bf v[4];   // [nt]
};

__global__ __launch_bounds__(256)
void attn_kernel(const __bf16* __restrict__ qg, const __bf16* __restrict__ kg,
                 const __bf16* __restrict__ vt, const float* __restrict__ x,
                 const float* __restrict__ emb, float* __restrict__ out) {
    __shared__ __bf16 pbuf[8][16 * 40];               // wave-private P staging

    const int lane  = threadIdx.x & 31;
    const int wave  = threadIdx.x >> 5;
    const int lhalf = lane >> 4;
    const int llo   = lane & 15;
    const int bh = blockIdx.x;                        // 0..31
    const int b  = bh / H_;
    const int h  = bh % H_;
    const int t0 = (blockIdx.y * 8 + wave) * 16;      // query tile within T

    const size_t qkbase = (size_t)b * T_ * D_ + (size_t)h * C_;
    const __bf16* kgl = kg + qkbase + (size_t)llo * D_ + lhalf * 16;
    const __bf16* vgl = vt + (size_t)(b * D_ + h * C_ + llo) * T_ + lhalf * 16;

    // Q fragments (C=64 -> two K=32 fragments), reused across all key tiles
    Frag16 qa[2];
    {
        const __bf16* qrow = qg + qkbase + (size_t)(t0 + llo) * D_;
#pragma unroll
        for (int kb = 0; kb < 2; ++kb) {
            qa[kb].q[0] = *(const uint4*)(qrow + kb * 32 + lhalf * 16);
            qa[kb].q[1] = *(const uint4*)(qrow + kb * 32 + 8 + lhalf * 16);
        }
    }

    float mrun[8], lrun[8];
    v8f o[4] = {v8zero(), v8zero(), v8zero(), v8zero()};
#pragma unroll
    for (int r = 0; r < 8; ++r) { mrun[r] = -INFINITY; lrun[r] = 0.f; }

    __bf16* pw = &pbuf[wave][0];
    const float scale = 0.125f;                       // 1/sqrt(C)

    auto load_kv = [&](int m0, KVFrags& f) {
        const __bf16* kr = kgl + (size_t)m0 * D_;
#pragma unroll
        for (int sub = 0; sub < 2; ++sub) {
#pragma unroll
            for (int kb = 0; kb < 2; ++kb)
                f.k[sub * 2 + kb] =
                    *(const v16bf*)(kr + (size_t)sub * 16 * D_ + kb * 32);
        }
        const __bf16* vr = vgl + m0;
#pragma unroll
        for (int nt = 0; nt < 4; ++nt)
            f.v[nt] = *(const v16bf*)(vr + (size_t)nt * 16 * T_);
    };

    auto step = [&](int m0, KVFrags& cur, int mnext, KVFrags& nxt) {
        // ---- S = Q * K^T for 32 keys (two 16-key sub-tiles) ----
        v8f s0 = v8zero(), s1 = v8zero();
        s0 = wmma_bf16(qa[0].v, cur.k[0], s0);
        s0 = wmma_bf16(qa[1].v, cur.k[1], s0);
        s1 = wmma_bf16(qa[0].v, cur.k[2], s1);
        s1 = wmma_bf16(qa[1].v, cur.k[3], s1);

        load_kv(mnext, nxt);                          // prefetch next key tile

        // ---- online softmax over the key (lane) dimension ----
        float corr[8];
#pragma unroll
        for (int r = 0; r < 8; ++r) {
            float a0 = s0[r] * scale;
            float a1 = s1[r] * scale;
            float tmax = redmax16(fmaxf(a0, a1));
            float mnew = fmaxf(mrun[r], tmax);
            float c2 = __expf(mrun[r] - mnew);
            float p0 = __expf(a0 - mnew);
            float p1 = __expf(a1 - mnew);
            lrun[r] = lrun[r] * c2 + redsum16(p0 + p1);
            mrun[r] = mnew;
            corr[r] = c2;
            // stage P (bf16) into LDS: row = query, col = key-in-tile
            const int row = r + lhalf * 8;
            pw[row * 40 + llo]      = (__bf16)p0;
            pw[row * 40 + 16 + llo] = (__bf16)p1;
        }

        // ---- re-layout: P A-fragment from LDS (row llo, K chunks) ----
        Frag16 pa;
        {
            const __bf16* pr = pw + llo * 40 + lhalf * 16;
            pa.q[0] = *(const uint4*)(pr);
            pa.q[1] = *(const uint4*)(pr + 8);
        }

        // ---- O = corr*O + P * V  (V^T layout: contiguous along keys) ----
#pragma unroll
        for (int nt = 0; nt < 4; ++nt) {
            v8f c = o[nt];
#pragma unroll
            for (int r = 0; r < 8; ++r) c[r] *= corr[r];
            o[nt] = wmma_bf16(pa.v, cur.v[nt], c);
        }
    };

    KVFrags fA, fB;
    load_kv(0, fA);
    for (int m0 = 0; m0 < T_; m0 += 64) {
        step(m0, fA, m0 + 32, fB);                       // m0+32 always < T
        step(m0 + 32, fB, (m0 + 64) & (T_ - 1), fA);     // wraps to 0 at the end
    }

    // ---- epilogue: y/l + x + emb ----
#pragma unroll
    for (int r = 0; r < 8; ++r) {
        const float inv = 1.0f / lrun[r];
        const int t = t0 + r + lhalf * 8;
        const size_t rowoff = ((size_t)b * T_ + t) * D_;
#pragma unroll
        for (int nt = 0; nt < 4; ++nt) {
            const int d = h * C_ + nt * 16 + llo;
            out[rowoff + d] = x[rowoff + d] + emb[b * D_ + d] + o[nt][r] * inv;
        }
    }
}

// ---------------------------------------------------------------------------
// Launcher
// ---------------------------------------------------------------------------
extern "C" void kernel_launch(void* const* d_in, const int* in_sizes, int n_in,
                              void* d_out, int out_size, void* d_ws, size_t ws_size,
                              hipStream_t stream) {
    const float* x     = (const float*)d_in[0];
    const float* emb   = (const float*)d_in[1];
    const float* Wq    = (const float*)d_in[2];
    const float* Wk    = (const float*)d_in[3];
    const float* Wv    = (const float*)d_in[4];
    const float* gamma = (const float*)d_in[5];
    const float* beta  = (const float*)d_in[6];
    float* out = (float*)d_out;

    // workspace layout (bytes)
    char* ws = (char*)d_ws;
    const size_t act = (size_t)B_ * T_ * D_ * sizeof(__bf16);   // 8 MB
    const size_t wsz = (size_t)D_ * D_ * sizeof(__bf16);        // 512 KB
    __bf16* xn  = (__bf16*)(ws);
    __bf16* qb  = (__bf16*)(ws + act);
    __bf16* kb  = (__bf16*)(ws + 2 * act);
    __bf16* vtb = (__bf16*)(ws + 3 * act);
    __bf16* wqb = (__bf16*)(ws + 4 * act);
    __bf16* wkb = (__bf16*)(ws + 4 * act + wsz);
    __bf16* wvb = (__bf16*)(ws + 4 * act + 2 * wsz);

    // 0) weights -> bf16
    castw_kernel<<<(D_ * D_) / 256, 256, 0, stream>>>(Wq, wqb);
    castw_kernel<<<(D_ * D_) / 256, 256, 0, stream>>>(Wk, wkb);
    castw_kernel<<<(D_ * D_) / 256, 256, 0, stream>>>(Wv, wvb);

    // 1) layernorm + cast
    ln_cast_kernel<<<(B_ * T_) / 8, 256, 0, stream>>>(x, gamma, beta, xn);

    // 2) QKV projections (V stored transposed)
    qkv_kernel<<<dim3((B_ * T_) / 16, 3), 256, 0, stream>>>(xn, wqb, wkb, wvb,
                                                            qb, kb, vtb);

    // 3) fused attention + residual
    attn_kernel<<<dim3(B_ * H_, T_ / 128), 256, 0, stream>>>(qb, kb, vtb, x, emb,
                                                             out);
}
